// SymbolicOperator_48593259987368
// MI455X (gfx1250) — compile-verified, matrix operand order checked by reading, not run on hardware
//
#include <hip/hip_runtime.h>
#include <hip/hip_bf16.h>

typedef __bf16 bf16_t;
typedef __attribute__((ext_vector_type(16))) __bf16 v16bf;
typedef __attribute__((ext_vector_type(8)))  float  v8f;

#define K_DIM   128
#define A_DIM   200
#define H_DIM   512      // K*NP1
#define H3      1536     // 3*H
#define B_TOT   256
#define L_DIM   32       // inner scan length (ACT-1)
#define WORDS   8
#define ROWS    16       // batch rows per workgroup (== WMMA M)
#define NWG     (B_TOT / ROWS)      // 16 workgroups
#define KTILES  (H_DIM / 32)        // 16 k-tiles of 32
#define NTILES  (H3 / 16)           // 96 n-tiles of 16
#define NT_PER_WAVE (NTILES / 8)    // 12 n-tiles per wave (8 waves/WG)

// ---- workspace layout ----
#define WF_BYTES   (NTILES * KTILES * 32 * 16 * 2)            // 1,572,864 B (bf16 fragments)
#define SV_OFFSET  WF_BYTES
#define SV_ELEMS   (B_TOT * L_DIM * A_DIM)                    // 1,638,400 f32

// ---- LDS layout (bytes) ----
#define SMEM_BYTES ( (ROWS*H_DIM + ROWS*H3 + L_DIM*K_DIM + K_DIM + \
                      2*ROWS*L_DIM + ROWS + 2*ROWS*A_DIM) * 4 + ROWS*H_DIM*2 )

// -----------------------------------------------------------------------------
// Pack w_hh (1536x512 f32, row-major) into WMMA B-matrix fragment order (bf16):
// Wf[nt][kt][lane][pos], 16 bf16 per lane = one contiguous 32B fragment slice.
// B[k][n] = w_hh[n][k];  lane = nn + 16*((kk>>3)&1);  pos = (kk&7) + 8*(kk>>4)
// -----------------------------------------------------------------------------
__global__ void pack_whh(const float* __restrict__ whh, bf16_t* __restrict__ Wf) {
    int idx = blockIdx.x * 256 + threadIdx.x;
    if (idx >= NTILES * KTILES * 32 * 16) return;
    int pos  =  idx        & 15;
    int lane = (idx >> 4)  & 31;
    int kt   = (idx >> 9)  & 15;
    int nt   =  idx >> 13;
    int nn = lane & 15;
    int kk = (pos & 7) + (((lane >> 4) & 1) << 3) + ((pos >> 3) << 4);
    int n = nt * 16 + nn;
    int k = kt * 32 + kk;
    Wf[idx] = (bf16_t)whh[n * H_DIM + k];
}

__global__ void init_sv(const float* __restrict__ isv, float* __restrict__ sv) {
    int idx = blockIdx.x * 256 + threadIdx.x;
    if (idx < SV_ELEMS) sv[idx] = isv[idx % A_DIM];
}

// -----------------------------------------------------------------------------
// Main recurrence: 16 WGs x 256 threads (8 waves). Each WG owns 16 batch rows.
// -----------------------------------------------------------------------------
__launch_bounds__(256)
__global__ void symop_main(const int*   __restrict__ instr,      // (8,256)
                           const float* __restrict__ prim_emb,   // (16,200)
                           const float* __restrict__ gate_w,     // (128)
                           const float* __restrict__ gate_b,     // (1)
                           const float* __restrict__ b_ih,       // (1536)
                           const float* __restrict__ b_hh,       // (1536)
                           const bf16_t* __restrict__ Wf,        // packed fragments
                           float* __restrict__ sv_g)             // (256,32,200)
{
    extern __shared__ char smem[];
    float* sh_h    = (float*)smem;                    // 16*512
    float* sh_gh   = sh_h    + ROWS * H_DIM;          // 16*1536
    float* sh_SK   = sh_gh   + ROWS * H3;             // 32*128
    float* sh_gw   = sh_SK   + L_DIM * K_DIM;         // 128
    float* sh_attr = sh_gw   + K_DIM;                 // 16*32  (read attn)
    float* sh_attw = sh_attr + ROWS * L_DIM;          // 16*32  (write attn)
    float* sh_gate = sh_attw + ROWS * L_DIM;          // 16
    float* sh_nv   = sh_gate + ROWS;                  // 16*200 (new_val)
    float* sh_prim = sh_nv   + ROWS * A_DIM;          // 16*200
    bf16_t* sh_hb  = (bf16_t*)(sh_prim + ROWS * A_DIM); // 16*512 bf16 (A operand)

    const int tid   = threadIdx.x;
    const int lane  = tid & 31;
    const int wv    = tid >> 5;
    const int brow0 = blockIdx.x * ROWS;

    // --- sinusoidal positional table SK = pe[:32], plus h0 = tile(pe[0],4) ---
    const float cdiv = 9.2103403719761836f / 128.0f;   // ln(10000)/K
    for (int i = tid; i < L_DIM * K_DIM; i += 256) {
        int l = i / K_DIM, d = i % K_DIM;
        float div = __expf(-cdiv * (float)(d & ~1));
        float arg = (float)l * div;
        sh_SK[i] = (d & 1) ? __cosf(arg) : __sinf(arg);
    }
    for (int i = tid; i < K_DIM; i += 256) sh_gw[i] = gate_w[i];
    for (int i = tid; i < ROWS * H_DIM; i += 256) {
        int d = i % K_DIM;                  // pe[0]: sin(0)=0 even, cos(0)=1 odd
        sh_h[i] = (d & 1) ? 1.0f : 0.0f;
    }
    __syncthreads();

    const float gb = gate_b[0];
    float* sv = sv_g + (size_t)brow0 * L_DIM * A_DIM;  // this WG's contiguous sv slab

    for (int w = 0; w < WORDS; ++w) {
        // primitive embedding gather for this word
        for (int i = tid; i < ROWS * A_DIM; i += 256) {
            int r = i / A_DIM, a = i % A_DIM;
            int word = instr[w * B_TOT + brow0 + r];
            sh_prim[i] = prim_emb[word * A_DIM + a];
        }
        __syncthreads();

        for (int step = 0; step < L_DIM; ++step) {
            // ---- stage A operand: h -> bf16 (shared by all 8 waves) ----
            for (int i = tid; i < ROWS * H_DIM; i += 256)
                sh_hb[i] = (bf16_t)sh_h[i];

            // ---- gate = sigmoid(h[:,3K:4K] @ gate_w + b) ----
            if (tid < ROWS) {
                float s = gb;
                const float* hr = sh_h + tid * H_DIM + 3 * K_DIM;
                for (int k = 0; k < K_DIM; ++k) s += hr[k] * sh_gw[k];
                sh_gate[tid] = 1.0f / (1.0f + __expf(-s));
            }
            // ---- attention scores: read_p/write_p @ SK.T (16x32 each, f32) ----
            for (int i = tid; i < ROWS * L_DIM; i += 256) {
                int r = i / L_DIM, l = i % L_DIM;
                const float* hr  = sh_h + r * H_DIM;
                const float* skl = sh_SK + l * K_DIM;
                float sr = 0.f, sw = 0.f;
                for (int k = 0; k < K_DIM; ++k) {
                    sr += hr[K_DIM     + k] * skl[k];
                    sw += hr[2 * K_DIM + k] * skl[k];
                }
                sh_attr[i] = sr;
                sh_attw[i] = sw;
            }
            __syncthreads();

            // ---- softmax over L (32 independent rows: 16 read + 16 write) ----
            if (tid < 2 * ROWS) {
                float* buf = (tid & 1) ? sh_attw : sh_attr;
                float* row = buf + (tid >> 1) * L_DIM;
                float mx = -1e30f;
                for (int l = 0; l < L_DIM; ++l) mx = fmaxf(mx, row[l]);
                float sum = 0.f;
                for (int l = 0; l < L_DIM; ++l) { float e = __expf(row[l] - mx); row[l] = e; sum += e; }
                float inv = 1.0f / sum;
                for (int l = 0; l < L_DIM; ++l) row[l] *= inv;
            }
            __syncthreads();

            // ---- read_val = attn . sv ; new_val = g*prim + (1-g)*read_val ----
            for (int i = tid; i < ROWS * A_DIM; i += 256) {
                int r = i / A_DIM, a = i % A_DIM;
                const float* svr = sv + (size_t)r * L_DIM * A_DIM + a;
                const float* at  = sh_attr + r * L_DIM;
                float rv = 0.f;
                for (int l = 0; l < L_DIM; ++l) rv += at[l] * svr[l * A_DIM];
                float g = sh_gate[r];
                sh_nv[i] = g * sh_prim[i] + (1.0f - g) * rv;
            }
            __syncthreads();

            // ---- sv blend: sv = wm*new_val + (1-wm)*sv (coalesced over a) ----
            for (int i = tid; i < ROWS * L_DIM * A_DIM; i += 256) {
                int a  = i % A_DIM;
                int rl = i / A_DIM;
                float wm = sh_attw[rl];                 // rl = r*32 + l
                int r = rl / L_DIM;
                sv[i] = wm * sh_nv[r * A_DIM + a] + (1.0f - wm) * sv[i];
            }
            __threadfence();

            // ---- WMMA GEMM: gh(16x1536) = h_bf16(16x512) x Wf ----
            v8f acc[NT_PER_WAVE];
            const v8f vzero = {0.f, 0.f, 0.f, 0.f, 0.f, 0.f, 0.f, 0.f};
            #pragma unroll
            for (int i = 0; i < NT_PER_WAVE; ++i) acc[i] = vzero;

            const int mm    = lane & 15;
            const int khalf = (lane >> 4) & 1;
            for (int kt = 0; kt < KTILES; ++kt) {
                v16bf afrag;
                const bf16_t* hrow = sh_hb + mm * H_DIM + kt * 32 + khalf * 8;
                #pragma unroll
                for (int p = 0; p < 16; ++p) {
                    int kk = (p & 7) + ((p >> 3) << 4);
                    afrag[p] = hrow[kk];
                }
                #pragma unroll
                for (int i = 0; i < NT_PER_WAVE; ++i) {
                    int nt = wv * NT_PER_WAVE + i;
                    const v16bf* bp =
                        (const v16bf*)(Wf + (size_t)(((nt * KTILES) + kt) * 32 + lane) * 16);
                    acc[i] = __builtin_amdgcn_wmma_f32_16x16x32_bf16(
                        false, afrag, false, *bp, (short)0, acc[i], false, false);
                }
            }
            // D layout: VGPR v, lanes 0-15 -> (M=v, N=lane); 16-31 -> (M=v+8, N=lane-16)
            {
                const int mbase = (lane >> 4) << 3;
                const int n     = lane & 15;
                #pragma unroll
                for (int i = 0; i < NT_PER_WAVE; ++i) {
                    int nt = wv * NT_PER_WAVE + i;
                    #pragma unroll
                    for (int v = 0; v < 8; ++v)
                        sh_gh[(mbase + v) * H3 + nt * 16 + n] = acc[i][v];
                }
            }
            __syncthreads();

            // ---- GRU update (gi = b_ih since x=0) ----
            for (int i = tid; i < ROWS * H_DIM; i += 256) {
                int r = i / H_DIM, k = i % H_DIM;
                const float* ghr = sh_gh + r * H3;
                float pr = b_ih[k]             + b_hh[k]             + ghr[k];
                float pz = b_ih[H_DIM + k]     + b_hh[H_DIM + k]     + ghr[H_DIM + k];
                float gn = b_hh[2 * H_DIM + k] + ghr[2 * H_DIM + k];
                float rr = 1.0f / (1.0f + __expf(-pr));
                float zz = 1.0f / (1.0f + __expf(-pz));
                float nn = tanhf(b_ih[2 * H_DIM + k] + rr * gn);
                sh_h[i] = (1.0f - zz) * nn + zz * sh_h[i];   // only this thread touches (r,k)
            }
            __syncthreads();
        }
    }
}

// -----------------------------------------------------------------------------
// Epilogue: actions = log_softmax(sv @ out_w.T + out_b) -> (B, 8, L),
// then padded_true.T = true_actions[1:].T as floats.
// -----------------------------------------------------------------------------
__global__ void epilogue(const float* __restrict__ sv_g,
                         const float* __restrict__ out_w,   // (8,200)
                         const float* __restrict__ out_b,   // (8)
                         const int*   __restrict__ trueA,   // (33,256)
                         float* __restrict__ out)
{
    int idx = blockIdx.x * 256 + threadIdx.x;   // over B*L = 8192
    if (idx >= B_TOT * L_DIM) return;
    int b = idx / L_DIM, l = idx % L_DIM;
    const float* svp = sv_g + (size_t)idx * A_DIM;

    float logit[8];
    float mx = -1e30f;
    #pragma unroll
    for (int o = 0; o < 8; ++o) {
        float s = out_b[o];
        const float* wrow = out_w + o * A_DIM;
        for (int a = 0; a < A_DIM; ++a) s += svp[a] * wrow[a];
        logit[o] = s;
        mx = fmaxf(mx, s);
    }
    float sum = 0.f;
    #pragma unroll
    for (int o = 0; o < 8; ++o) sum += __expf(logit[o] - mx);
    float lse = mx + __logf(sum);
    #pragma unroll
    for (int o = 0; o < 8; ++o)
        out[(b * 8 + o) * L_DIM + l] = logit[o] - lse;   // (B,8,L)

    out[B_TOT * 8 * L_DIM + b * L_DIM + l] = (float)trueA[(l + 1) * B_TOT + b];
}

extern "C" void kernel_launch(void* const* d_in, const int* in_sizes, int n_in,
                              void* d_out, int out_size, void* d_ws, size_t ws_size,
                              hipStream_t stream) {
    (void)in_sizes; (void)n_in; (void)out_size; (void)ws_size;
    const int*   instr = (const int*)d_in[0];
    const int*   trueA = (const int*)d_in[1];
    const float* isv   = (const float*)d_in[2];
    // d_in[3] program_emb: unused by the reference math
    const float* prim  = (const float*)d_in[4];
    const float* gw    = (const float*)d_in[5];
    const float* gb    = (const float*)d_in[6];
    // d_in[7] w_ih: multiplied by zeros in the reference -> unused
    const float* whh   = (const float*)d_in[8];
    const float* bih   = (const float*)d_in[9];
    const float* bhh   = (const float*)d_in[10];
    const float* outw  = (const float*)d_in[11];
    const float* outb  = (const float*)d_in[12];

    bf16_t* Wf = (bf16_t*)d_ws;
    float*  sv = (float*)((char*)d_ws + SV_OFFSET);

    pack_whh<<<(NTILES * KTILES * 32 * 16) / 256, 256, 0, stream>>>(whh, Wf);
    init_sv<<<(SV_ELEMS + 255) / 256, 256, 0, stream>>>(isv, sv);
    symop_main<<<NWG, 256, SMEM_BYTES, stream>>>(instr, prim, gw, gb, bih, bhh, Wf, sv);
    epilogue<<<(B_TOT * L_DIM + 255) / 256, 256, 0, stream>>>(sv, outw, outb, trueA,
                                                              (float*)d_out);
}
